// GroverTransEncoder_33947421507740
// MI455X (gfx1250) — matrix operand order repaired and checked by reading, not compiled, not run on hardware
//
#include <hip/hip_runtime.h>

// ---------------- problem constants (from reference setup_inputs) -----------
constexpr int H      = 256;   // hidden
constexpr int FF     = 1024;  // ffn inner
constexpr int L      = 4;     // layers
constexpr int NHEAD  = 8;
constexpr int DK     = 32;    // H / NHEAD
constexpr int G      = 512;   // num_graphs
constexpr int MAXLEN = 112;

typedef __attribute__((ext_vector_type(16))) __bf16 v16bf;
typedef __attribute__((ext_vector_type(8)))  __bf16 v8bf;
typedef __attribute__((ext_vector_type(8)))  float  v8f;
typedef __attribute__((ext_vector_type(4)))  int    v4i;

__device__ __forceinline__ __bf16 f2bf(float f) {
  union { float f; unsigned u; } v; v.f = f;
  unsigned r = (v.u + 0x7FFFu + ((v.u >> 16) & 1u)) >> 16;  // RNE
  unsigned short s = (unsigned short)r;
  __bf16 b;
  __builtin_memcpy(&b, &s, sizeof(b));
  return b;
}

// ---- CDNA5 async global->LDS copy (ASYNCcnt path), with safe fallback ------
#if __has_builtin(__builtin_amdgcn_global_load_async_to_lds_b128)
#define ASYNC_LDS 1
#else
#define ASYNC_LDS 0
#endif

typedef __attribute__((address_space(1))) v4i gv4i;  // global int4
typedef __attribute__((address_space(3))) v4i lv4i;  // LDS int4

__device__ __forceinline__ void async_copy16B(void* lds, const void* g) {
#if ASYNC_LDS
  __builtin_amdgcn_global_load_async_to_lds_b128((gv4i*)g, (lv4i*)lds, 0, 0);
#else
  *(v8bf*)lds = *(const v8bf*)g;
#endif
}
__device__ __forceinline__ void async_wait_all() {
#if ASYNC_LDS
#if __has_builtin(__builtin_amdgcn_s_wait_asynccnt)
  __builtin_amdgcn_s_wait_asynccnt(0);
#else
  asm volatile("s_wait_asynccnt 0x0" ::: "memory");
#endif
#endif
}

// ---------------- tiled WMMA bf16 GEMM --------------------------------------
// A:  [M,K] bf16 row-major (activations)
// Bt: [N,K] bf16 row-major (pre-transposed weights, fragments read directly)
// C = act(A@B + bias) (+resid);  out fp32 (Cf) or bf16 (Cb) per template.
// Block = 256 threads = 8 waves; macro-tile 128(M) x 64(N), K-step 32.
// Each wave owns a 32x32 sub-tile = 2x2 WMMA 16x16x32 tiles.
// Requires: M % 128 == 0, N % 64 == 0, K % 32 == 0 (true at all call sites).
template <bool RELU, bool RESID, bool OUTB>
__global__ __launch_bounds__(256)
void gemm_wmma_bf16(const __bf16* __restrict__ A, const __bf16* __restrict__ Bt,
                    const float* __restrict__ bias, const float* __restrict__ resid,
                    float* __restrict__ Cf, __bf16* __restrict__ Cb,
                    int M, int N, int K)
{
  constexpr int BM = 128, BN = 64, BK = 32;
  constexpr int LDT = BK + 8;                 // bf16 row stride (80B = 5*16B)
  __shared__ __bf16 As[BM][LDT];              // A panel, row-major [m][k]

  const int tid  = threadIdx.x;
  const int wave = tid >> 5;
  const int lane = tid & 31;
  const int lo = lane & 15, hi = lane >> 4;
  const int wm = (wave & 3) * 32;             // wave M offset within block tile
  const int wn = (wave >> 2) * 32;            // wave N offset within block tile

  const int nTilesN = N / BN;
  const int bM = (blockIdx.x / nTilesN) * BM;
  const int bN = (blockIdx.x % nTilesN) * BN;

  // A staging map: thread t copies 32B (16 bf16) = two async 16B chunks
  const int arow = tid >> 1;                  // 0..127
  const int ak   = (tid & 1) * 16;            // 0 or 16
  const __bf16* __restrict__ aSrc = A + (size_t)(bM + arow) * K + ak;

  v8f acc[2][2] = {};

  for (int k0 = 0; k0 < K; k0 += BK) {
    // ---- A panel: pure bf16 copy global -> LDS via async path ----
    async_copy16B(&As[arow][ak], aSrc + k0);
    async_copy16B(&As[arow][ak + 8], aSrc + k0 + 8);
    async_wait_all();
    __syncthreads();

    // ---- fragments ----
    v16bf afrag[2], bfrag[2];
#pragma unroll
    for (int ti = 0; ti < 2; ++ti) {
      // A 16x32 fragment: lane holds row (wm+ti*16+lo); 16 contiguous k at
      // k = hi*16 (matches B fragment order).
      const __bf16* pa = &As[wm + ti * 16 + lo][hi * 16];
      v8bf a0 = *(const v8bf*)pa;
      v8bf a1 = *(const v8bf*)(pa + 8);
      afrag[ti] = __builtin_shufflevector(a0, a1, 0,1,2,3,4,5,6,7,8,9,10,11,12,13,14,15);
    }
#pragma unroll
    for (int tj = 0; tj < 2; ++tj) {
      // B 32x16 fragment: lane holds col (wn+tj*16+lo); 16 contiguous k at
      // k = hi*16, read straight from pre-transposed bf16 weights (L2-hot).
      const __bf16* pb = Bt + (size_t)(bN + wn + tj * 16 + lo) * K + k0 + hi * 16;
      v8bf b0 = *(const v8bf*)pb;
      v8bf b1 = *(const v8bf*)(pb + 8);
      bfrag[tj] = __builtin_shufflevector(b0, b1, 0,1,2,3,4,5,6,7,8,9,10,11,12,13,14,15);
    }
#pragma unroll
    for (int ti = 0; ti < 2; ++ti)
#pragma unroll
      for (int tj = 0; tj < 2; ++tj)
        acc[ti][tj] = __builtin_amdgcn_wmma_f32_16x16x32_bf16(
            false, afrag[ti], false, bfrag[tj], (short)0, acc[ti][tj], false, false);
    __syncthreads();
  }

  // ---- branch-free epilogue ----
#pragma unroll
  for (int ti = 0; ti < 2; ++ti) {
#pragma unroll
    for (int tj = 0; tj < 2; ++tj) {
      const int col = bN + wn + tj * 16 + lo;
      const float bval = bias[col];
#pragma unroll
      for (int r = 0; r < 8; ++r) {
        const int row = bM + wm + ti * 16 + r + 8 * hi;
        const size_t idx = (size_t)row * N + col;
        float val = acc[ti][tj][r] + bval;
        if (RELU)  val = val > 0.f ? val : 0.f;
        if (RESID) val += resid[idx];
        if (OUTB)  Cb[idx] = f2bf(val);
        else       Cf[idx] = val;
      }
    }
  }
}

// ---------------- weight prep: fp32 [K,N] -> bf16 transposed [N,K] ----------
__global__ void wtrans_kernel(const float* __restrict__ W, __bf16* __restrict__ Bt,
                              int K, int N) {
  int idx = blockIdx.x * blockDim.x + threadIdx.x;
  if (idx < K * N) {
    int k = idx / N, n = idx % N;
    Bt[(size_t)n * K + k] = f2bf(W[idx]);
  }
}

// ---------------- small utility kernels -------------------------------------
__global__ void fill_f(float* p, float v, size_t n) {
  size_t i = (size_t)blockIdx.x * blockDim.x + threadIdx.x;
  if (i < n) p[i] = v;
}
__global__ void fill_i(int* p, int v, size_t n) {
  size_t i = (size_t)blockIdx.x * blockDim.x + threadIdx.x;
  if (i < n) p[i] = v;
}
__global__ void fill_h(__bf16* p, size_t n) {
  size_t i = (size_t)blockIdx.x * blockDim.x + threadIdx.x;
  if (i < n) p[i] = (__bf16)0.0f;
}
__global__ void copy_f(float* dst, const float* src, size_t n) {
  size_t i = (size_t)blockIdx.x * blockDim.x + threadIdx.x;
  if (i < n) dst[i] = src[i];
}

// degree over destination (col) nodes
__global__ void deg_kernel(const int* __restrict__ ei, float* __restrict__ deg, int E) {
  int e = blockIdx.x * blockDim.x + threadIdx.x;
  if (e < E) atomicAdd(&deg[ei[E + e]], 1.0f);
}
__global__ void clamp_deg(float* deg, int N) {
  int i = blockIdx.x * blockDim.x + threadIdx.x;
  if (i < N) deg[i] = fmaxf(deg[i], 1.0f);
}

// agg[col] += x[row] : one block (256 threads = H features) per edge
__global__ __launch_bounds__(256)
void scatter_kernel(const float* __restrict__ x, const int* __restrict__ ei,
                    float* __restrict__ agg, int E) {
  const int e = blockIdx.x;
  const int f = threadIdx.x;
  const int r = ei[e];
  const int c = ei[E + e];
  atomicAdd(&agg[(size_t)c * H + f], x[(size_t)r * H + f]);
}

// aggbf = bf16(agg / deg)
__global__ __launch_bounds__(256)
void scale_rows_bf(const float* __restrict__ agg, const float* __restrict__ deg,
                   __bf16* __restrict__ aggbf) {
  const int i = blockIdx.x;
  const int f = threadIdx.x;
  aggbf[(size_t)i * H + f] = f2bf(agg[(size_t)i * H + f] / deg[i]);
}

__global__ void count_kernel(const int* __restrict__ batch, int* __restrict__ counts, int N) {
  int i = blockIdx.x * blockDim.x + threadIdx.x;
  if (i < N) atomicAdd(&counts[batch[i]], 1);
}

// exclusive prefix sum over G=512 counts, single block
__global__ __launch_bounds__(G)
void scan_kernel(const int* __restrict__ counts, int* __restrict__ offsets) {
  __shared__ int s[G];
  const int i = threadIdx.x;
  s[i] = counts[i];
  __syncthreads();
  for (int st = 1; st < G; st <<= 1) {
    int v = (i >= st) ? s[i - st] : 0;
    __syncthreads();
    s[i] += v;
    __syncthreads();
  }
  offsets[i] = s[i] - counts[i];
}

__global__ void pos_kernel(const int* __restrict__ batch, const int* __restrict__ offsets,
                           int* __restrict__ pos, int N) {
  int i = blockIdx.x * blockDim.x + threadIdx.x;
  if (i < N) pos[i] = i - offsets[batch[i]];
}

// pack fp32 node states into padded bf16 [G, MAXLEN, H]
__global__ __launch_bounds__(256)
void pack_kernel(const float* __restrict__ x, const int* __restrict__ batch,
                 const int* __restrict__ pos, __bf16* __restrict__ hbuf,
                 int* __restrict__ mask) {
  const int i = blockIdx.x;
  const int f = threadIdx.x;
  const int g = batch[i], p = pos[i];
  hbuf[((size_t)g * MAXLEN + p) * H + f] = f2bf(x[(size_t)i * H + f]);
  if (f == 0) mask[g * MAXLEN + p] = 1;
}

__global__ __launch_bounds__(256)
void unpad_kernel(const __bf16* __restrict__ hbuf, const int* __restrict__ batch,
                  const int* __restrict__ pos, float* __restrict__ out) {
  const int i = blockIdx.x;
  const int f = threadIdx.x;
  const int g = batch[i], p = pos[i];
  out[(size_t)i * H + f] = (float)hbuf[((size_t)g * MAXLEN + p) * H + f];
}

// ---------------- attention: one block per (graph, head) --------------------
__global__ __launch_bounds__(128)
void attn_kernel(const __bf16* __restrict__ Q, const __bf16* __restrict__ Km,
                 const __bf16* __restrict__ Vm, const int* __restrict__ mask,
                 __bf16* __restrict__ Out) {
  const int g  = blockIdx.x;
  const int hd = blockIdx.y;
  __shared__ float Ks[MAXLEN][DK];
  __shared__ float Vs[MAXLEN][DK];
  const int base = g * MAXLEN;
  for (int idx = threadIdx.x; idx < MAXLEN * DK; idx += blockDim.x) {
    const int t = idx / DK, d = idx % DK;
    Ks[t][d] = (float)Km[(size_t)(base + t) * H + hd * DK + d];
    Vs[t][d] = (float)Vm[(size_t)(base + t) * H + hd * DK + d];
  }
  __syncthreads();
  const float scale = 0.1767766952966369f;  // 1/sqrt(32)
  for (int q = threadIdx.x; q < MAXLEN; q += blockDim.x) {
    float qv[DK];
#pragma unroll
    for (int d = 0; d < DK; ++d) qv[d] = (float)Q[(size_t)(base + q) * H + hd * DK + d];
    float s[MAXLEN];
    float mx = -1e30f;
    for (int j = 0; j < MAXLEN; ++j) {
      float acc = 0.f;
#pragma unroll
      for (int d = 0; d < DK; ++d) acc += qv[d] * Ks[j][d];
      acc *= scale;
      if (!mask[base + j]) acc = -1e9f;
      s[j] = acc;
      mx = fmaxf(mx, acc);
    }
    float sum = 0.f;
    for (int j = 0; j < MAXLEN; ++j) { float e = __expf(s[j] - mx); s[j] = e; sum += e; }
    const float inv = 1.f / sum;
    float o[DK];
#pragma unroll
    for (int d = 0; d < DK; ++d) o[d] = 0.f;
    for (int j = 0; j < MAXLEN; ++j) {
      const float w = s[j] * inv;
#pragma unroll
      for (int d = 0; d < DK; ++d) o[d] += w * Vs[j][d];
    }
#pragma unroll
    for (int d = 0; d < DK; ++d) Out[(size_t)(base + q) * H + hd * DK + d] = f2bf(o[d]);
  }
}

// ---------------- fused residual + LayerNorm (fp32 math, bf16 storage) ------
__global__ __launch_bounds__(256)
void ln_residual_kernel(__bf16* __restrict__ hbuf, const __bf16* __restrict__ add,
                        const float* __restrict__ g, const float* __restrict__ b) {
  const int t = blockIdx.x;
  const int f = threadIdx.x;  // == H
  const float z = (float)hbuf[(size_t)t * H + f] + (float)add[(size_t)t * H + f];
  __shared__ float red[H];
  red[f] = z;
  __syncthreads();
  for (int st = H / 2; st > 0; st >>= 1) { if (f < st) red[f] += red[f + st]; __syncthreads(); }
  const float mu = red[0] / H;
  __syncthreads();
  const float d = z - mu;
  red[f] = d * d;
  __syncthreads();
  for (int st = H / 2; st > 0; st >>= 1) { if (f < st) red[f] += red[f + st]; __syncthreads(); }
  const float var = red[0] / H;
  const float r = rsqrtf(var + 1e-5f);
  hbuf[(size_t)t * H + f] = f2bf(d * r * g[f] + b[f]);
}

// ---------------- host-side orchestration -----------------------------------
static inline unsigned cdiv(size_t a, size_t b) { return (unsigned)((a + b - 1) / b); }

extern "C" void kernel_launch(void* const* d_in, const int* in_sizes, int n_in,
                              void* d_out, int out_size, void* d_ws, size_t ws_size,
                              hipStream_t stream) {
  (void)n_in; (void)out_size; (void)ws_size;
  const float* x     = (const float*)d_in[0];
  const int*   ei    = (const int*)  d_in[1];
  const int*   batch = (const int*)  d_in[3];
  const float* Wh    = (const float*)d_in[6];
  const float* bh    = (const float*)d_in[7];
  const float* Wq    = (const float*)d_in[8];
  const float* bq    = (const float*)d_in[9];
  const float* Wk    = (const float*)d_in[10];
  const float* bk    = (const float*)d_in[11];
  const float* Wv    = (const float*)d_in[12];
  const float* bv    = (const float*)d_in[13];
  const float* Wo    = (const float*)d_in[14];
  const float* bo    = (const float*)d_in[15];
  const float* W1    = (const float*)d_in[16];
  const float* b1    = (const float*)d_in[17];
  const float* W2    = (const float*)d_in[18];
  const float* b2    = (const float*)d_in[19];
  const float* ln1g  = (const float*)d_in[20];
  const float* ln1b  = (const float*)d_in[21];
  const float* ln2g  = (const float*)d_in[22];
  const float* ln2b  = (const float*)d_in[23];

  const int N = in_sizes[0] / H;  // 32768
  const int E = in_sizes[1] / 2;  // 131072
  const int T = G * MAXLEN;       // 57344 (multiple of 128)

  // ---- workspace carve-out (256B aligned slices) ----
  char* ws = (char*)d_ws;
  size_t off = 0;
  auto carve = [&](size_t bytes) -> char* {
    char* p = ws + off;
    off += (bytes + 255) & ~(size_t)255;
    return p;
  };
  float*  xbuf    = (float*) carve((size_t)N * H * 4);
  float*  agg     = (float*) carve((size_t)N * H * 4);
  __bf16* aggbf   = (__bf16*)carve((size_t)N * H * 2);
  float*  deg     = (float*) carve((size_t)N * 4);
  int*    counts  = (int*)   carve((size_t)G * 4);
  int*    offsets = (int*)   carve((size_t)G * 4);
  int*    pos     = (int*)   carve((size_t)N * 4);
  int*    mask    = (int*)   carve((size_t)T * 4);
  __bf16* hbuf    = (__bf16*)carve((size_t)T * H * 2);
  // Q|K|V|AO contiguous: 4*T*H bf16 == T*FF bf16, reused as FFN hidden
  __bf16* Qb      = (__bf16*)carve((size_t)T * H * 2);
  __bf16* Kb      = (__bf16*)carve((size_t)T * H * 2);
  __bf16* Vb      = (__bf16*)carve((size_t)T * H * 2);
  __bf16* AOb     = (__bf16*)carve((size_t)T * H * 2);
  __bf16* F2b     = (__bf16*)carve((size_t)T * H * 2);
  __bf16* F1b     = Qb;  // [T, FF] overlay (Q/K/V/AO dead during FFN)
  // pre-transposed bf16 weights [N,K]
  __bf16* WhT     = (__bf16*)carve((size_t)H * H * 2);
  __bf16* WqT     = (__bf16*)carve((size_t)L * H * H * 2);
  __bf16* WkT     = (__bf16*)carve((size_t)L * H * H * 2);
  __bf16* WvT     = (__bf16*)carve((size_t)L * H * H * 2);
  __bf16* WoT     = (__bf16*)carve((size_t)L * H * H * 2);
  __bf16* W1T     = (__bf16*)carve((size_t)L * H * FF * 2);
  __bf16* W2T     = (__bf16*)carve((size_t)L * H * FF * 2);

  const size_t NH_elems = (size_t)N * H;
  const size_t TH_elems = (size_t)T * H;

  // ---- weight prep (runs every call; deterministic) ----
  const unsigned wtHH = cdiv((size_t)H * H, 256);
  const unsigned wtHF = cdiv((size_t)H * FF, 256);
  wtrans_kernel<<<wtHH, 256, 0, stream>>>(Wh, WhT, H, H);
  for (int l = 0; l < L; ++l) {
    wtrans_kernel<<<wtHH, 256, 0, stream>>>(Wq + (size_t)l * H * H, WqT + (size_t)l * H * H, H, H);
    wtrans_kernel<<<wtHH, 256, 0, stream>>>(Wk + (size_t)l * H * H, WkT + (size_t)l * H * H, H, H);
    wtrans_kernel<<<wtHH, 256, 0, stream>>>(Wv + (size_t)l * H * H, WvT + (size_t)l * H * H, H, H);
    wtrans_kernel<<<wtHH, 256, 0, stream>>>(Wo + (size_t)l * H * H, WoT + (size_t)l * H * H, H, H);
    wtrans_kernel<<<wtHF, 256, 0, stream>>>(W1 + (size_t)l * H * FF, W1T + (size_t)l * FF * H, H, FF);
    wtrans_kernel<<<wtHF, 256, 0, stream>>>(W2 + (size_t)l * FF * H, W2T + (size_t)l * H * FF, FF, H);
  }

  // ---- MPN atom encoder ----
  copy_f<<<cdiv(NH_elems, 256), 256, 0, stream>>>(xbuf, x, NH_elems);
  fill_f<<<cdiv(N, 256), 256, 0, stream>>>(deg, 0.f, (size_t)N);
  deg_kernel<<<cdiv(E, 256), 256, 0, stream>>>(ei, deg, E);
  clamp_deg<<<cdiv(N, 256), 256, 0, stream>>>(deg, N);

  const unsigned mpnBlocks = (unsigned)((N / 128) * (H / 64));
  for (int l = 0; l < L; ++l) {
    fill_f<<<cdiv(NH_elems, 256), 256, 0, stream>>>(agg, 0.f, NH_elems);
    scatter_kernel<<<E, 256, 0, stream>>>(xbuf, ei, agg, E);
    scale_rows_bf<<<N, 256, 0, stream>>>(agg, deg, aggbf);
    // x = x + relu(agg/deg @ Wh + bh)   (fp32 out, fp32 residual)
    gemm_wmma_bf16<true, true, false><<<mpnBlocks, 256, 0, stream>>>(
        aggbf, WhT, bh, xbuf, xbuf, nullptr, N, H, H);
  }

  // ---- pack ragged graphs into padded bf16 [G, MAXLEN, H] ----
  fill_i<<<cdiv(G, 256), 256, 0, stream>>>(counts, 0, (size_t)G);
  count_kernel<<<cdiv(N, 256), 256, 0, stream>>>(batch, counts, N);
  scan_kernel<<<1, G, 0, stream>>>(counts, offsets);
  pos_kernel<<<cdiv(N, 256), 256, 0, stream>>>(batch, offsets, pos, N);
  fill_h<<<cdiv(TH_elems, 256), 256, 0, stream>>>(hbuf, TH_elems);
  fill_i<<<cdiv(T, 256), 256, 0, stream>>>(mask, 0, (size_t)T);
  pack_kernel<<<N, 256, 0, stream>>>(xbuf, batch, pos, hbuf, mask);

  // ---- transformer layers ----
  const unsigned projBlocks = (unsigned)((T / 128) * (H / 64));
  const unsigned ffn1Blocks = (unsigned)((T / 128) * (FF / 64));
  for (int l = 0; l < L; ++l) {
    const __bf16* WqTl = WqT + (size_t)l * H * H;  const float* bql = bq + (size_t)l * H;
    const __bf16* WkTl = WkT + (size_t)l * H * H;  const float* bkl = bk + (size_t)l * H;
    const __bf16* WvTl = WvT + (size_t)l * H * H;  const float* bvl = bv + (size_t)l * H;
    const __bf16* WoTl = WoT + (size_t)l * H * H;  const float* bol = bo + (size_t)l * H;
    const __bf16* W1Tl = W1T + (size_t)l * FF * H; const float* b1l = b1 + (size_t)l * FF;
    const __bf16* W2Tl = W2T + (size_t)l * H * FF; const float* b2l = b2 + (size_t)l * H;

    gemm_wmma_bf16<false, false, true><<<projBlocks, 256, 0, stream>>>(
        hbuf, WqTl, bql, nullptr, nullptr, Qb, T, H, H);
    gemm_wmma_bf16<false, false, true><<<projBlocks, 256, 0, stream>>>(
        hbuf, WkTl, bkl, nullptr, nullptr, Kb, T, H, H);
    gemm_wmma_bf16<false, false, true><<<projBlocks, 256, 0, stream>>>(
        hbuf, WvTl, bvl, nullptr, nullptr, Vb, T, H, H);

    attn_kernel<<<dim3(G, NHEAD), 128, 0, stream>>>(Qb, Kb, Vb, mask, AOb);

    // o = attn_out @ Wo + bo  (Q buffer dead, reuse as output)
    gemm_wmma_bf16<false, false, true><<<projBlocks, 256, 0, stream>>>(
        AOb, WoTl, bol, nullptr, nullptr, Qb, T, H, H);
    ln_residual_kernel<<<T, H, 0, stream>>>(hbuf, Qb,
                                            ln1g + (size_t)l * H, ln1b + (size_t)l * H);

    // FFN (F1b overlays Q|K|V|AO which are all dead now)
    gemm_wmma_bf16<true, false, true><<<ffn1Blocks, 256, 0, stream>>>(
        hbuf, W1Tl, b1l, nullptr, nullptr, F1b, T, FF, H);
    gemm_wmma_bf16<false, false, true><<<projBlocks, 256, 0, stream>>>(
        F1b, W2Tl, b2l, nullptr, nullptr, F2b, T, H, FF);
    ln_residual_kernel<<<T, H, 0, stream>>>(hbuf, F2b,
                                            ln2g + (size_t)l * H, ln2b + (size_t)l * H);
  }

  // ---- unpad back to node order ----
  unpad_kernel<<<N, 256, 0, stream>>>(hbuf, batch, pos, (float*)d_out);
}
